// NextBasketEncoder_71408126263376
// MI455X (gfx1250) — compile-verified, matrix operand order checked by reading, not run on hardware
//
#include <hip/hip_runtime.h>
#include <hip/hip_bf16.h>
#include <cstdint>
#include <cstddef>

// ---------------------------------------------------------------------------
// CDNA5 (gfx1250) WMMA helpers: v_wmma_f32_16x16x32_bf16, wave32.
// A operand: lane l holds row m=(l&15), K chunks [kb..kb+7] and [kb+16..kb+23],
//            kb=(l>>4)*8  (CDNA5 ISA "16-bit A-Matrix 16x32").
// B operand: identical pattern with rows = columns of B = rows of W[N][K].
// C/D:       vgpr r holds (m = r + 8*(l>>4), n = l&15).
// ---------------------------------------------------------------------------
typedef __attribute__((ext_vector_type(16))) __bf16 v16bf;
typedef __attribute__((ext_vector_type(8)))  __bf16 v8bf;
typedef __attribute__((ext_vector_type(4)))  __bf16 v4bf;
typedef __attribute__((ext_vector_type(8)))  float  v8f;
typedef __attribute__((ext_vector_type(4)))  float  v4f;

#define DEV_INLINE __device__ __forceinline__

DEV_INLINE v16bf load_frag(const __bf16* __restrict__ base, int row0, int ld, int k0) {
  const int lane = threadIdx.x & 31;
  const __bf16* p = base + (size_t)(row0 + (lane & 15)) * (size_t)ld
                         + (size_t)(k0 + ((lane >> 4) << 3));
  v8bf lo = *(const v8bf*)p;
  v8bf hi = *(const v8bf*)(p + 16);
  v16bf f;
#pragma unroll
  for (int i = 0; i < 8; ++i) { f[i] = lo[i]; f[i + 8] = hi[i]; }
  return f;
}

DEV_INLINE v8f wmma_bf(v16bf a, v16bf b, v8f c) {
  return __builtin_amdgcn_wmma_f32_16x16x32_bf16(false, a, false, b, (short)0, c,
                                                 false, false);
}

DEV_INLINE float sigmoidf_(float x) { return 1.0f / (1.0f + __expf(-x)); }

// ---------------------------------------------------------------------------
// Utility kernels
// ---------------------------------------------------------------------------
__global__ void cvt_f32_bf16_kernel(const float* __restrict__ src,
                                    __bf16* __restrict__ dst, int n4) {
  int i = blockIdx.x * 256 + threadIdx.x;       // one float4 per thread
  if (i < n4) {
    v4f x = *(const v4f*)(src + (size_t)i * 4);
    v4bf y;
#pragma unroll
    for (int j = 0; j < 4; ++j) y[j] = (__bf16)x[j];
    *(v4bf*)(dst + (size_t)i * 4) = y;
  }
}

__global__ void zero_u32_kernel(uint32_t* __restrict__ p, int n) {
  int i = blockIdx.x * 256 + threadIdx.x;
  if (i < n) p[i] = 0u;
}

// ---------------------------------------------------------------------------
// Kernel 1: embedding gather + per-order item self-attention + token mean.
// One block per (user, order) pair: 32 tokens x 128 dims, 4 heads (hd=32).
// Everything stays in LDS; v stored transposed so P@V B-operand is contiguous.
// ---------------------------------------------------------------------------
__global__ __launch_bounds__(256) void attn_pool_kernel(
    const int* __restrict__ hist_all,       // [8192*32]
    const float* __restrict__ emb,          // [(V+1)*128]
    const __bf16* __restrict__ w_in,        // [384*128]
    const float* __restrict__ b_in,         // [384]
    const __bf16* __restrict__ w_out,       // [128*128]
    const float* __restrict__ b_out,        // [128]
    float* __restrict__ oe_f,               // [8192*128]
    __bf16* __restrict__ oe_bf)             // [8192*128]
{
  __shared__ __align__(16) char smem[57344];
  __bf16* xs  = (__bf16*)(smem);           // [32][128] x (aliased by os later)
  __bf16* qs  = (__bf16*)(smem +  8192);   // [32][128]
  __bf16* ks  = (__bf16*)(smem + 16384);   // [32][128]
  __bf16* vts = (__bf16*)(smem + 24576);   // [128][32] transposed
  float*  sc  = (float*) (smem + 32768);   // [4][32][32] (aliased by ys later)
  __bf16* ps  = (__bf16*)(smem + 49152);   // [4][32][32]
  __bf16* os  = (__bf16*)(smem);           // alias xs
  float*  ys  = (float*) (smem + 32768);   // alias sc

  const int bo   = blockIdx.x;
  const int tid  = threadIdx.x;
  const int wave = tid >> 5;
  const int lane = tid & 31;
  const int* hist = hist_all + bo * 32;

  // vectorized gather: float4 load -> packed bf16x4 LDS store
  for (int i = tid; i < 32 * 32; i += 256) {
    int t = i >> 5, d4 = (i & 31) << 2;
    v4f x = *(const v4f*)(emb + (size_t)hist[t] * 128 + d4);
    v4bf y;
#pragma unroll
    for (int j = 0; j < 4; ++j) y[j] = (__bf16)x[j];
    *(v4bf*)(xs + t * 128 + d4) = y;
  }
  __syncthreads();

  // qkv = x @ w_in^T + b_in   (M=32,N=384,K=128): 48 tiles, 6 per wave
  for (int i = 0; i < 6; ++i) {
    int idx = wave * 6 + i;
    int nt = idx % 24, mt = idx / 24;
    int n0 = nt * 16, m0 = mt * 16;
    v8f acc = {};
    for (int k0 = 0; k0 < 128; k0 += 32)
      acc = wmma_bf(load_frag(xs, m0, 128, k0), load_frag(w_in, n0, 128, k0), acc);
    int n = n0 + (lane & 15);
    int m = m0 + ((lane >> 4) << 3);
    float bias = b_in[n];
    int region = __builtin_amdgcn_readfirstlane(n0);   // wave-uniform branch
    if (region < 128) {            // q, row-major
#pragma unroll
      for (int r = 0; r < 8; ++r) qs[(m + r) * 128 + n] = (__bf16)(acc[r] + bias);
    } else if (region < 256) {     // k, row-major
#pragma unroll
      for (int r = 0; r < 8; ++r) ks[(m + r) * 128 + (n - 128)] = (__bf16)(acc[r] + bias);
    } else {                       // v, transposed: contiguous per lane -> b128
      v8bf pk;
#pragma unroll
      for (int r = 0; r < 8; ++r) pk[r] = (__bf16)(acc[r] + bias);
      *(v8bf*)(vts + (n - 256) * 32 + m) = pk;
    }
  }
  __syncthreads();

  // scores = q_h @ k_h^T / sqrt(32): per head 2x2 tiles, K=32. 16 tiles total.
  for (int i = 0; i < 2; ++i) {
    int idx = wave * 2 + i;
    int h = idx >> 2, mt = (idx >> 1) & 1, nt = idx & 1;
    v8f acc = {};
    acc = wmma_bf(load_frag(qs + h * 32, mt * 16, 128, 0),
                  load_frag(ks + h * 32, nt * 16, 128, 0), acc);
    int n = nt * 16 + (lane & 15);
    int m = mt * 16 + ((lane >> 4) << 3);
#pragma unroll
    for (int r = 0; r < 8; ++r)
      sc[h * 1024 + (m + r) * 32 + n] = acc[r] * 0.17677669529663687f;
  }
  __syncthreads();

  // softmax over 32 keys; 128 rows (= 4 heads * 32 queries)
  if (tid < 128) {
    const float* row = sc + tid * 32;
    float mx = row[0];
#pragma unroll
    for (int k = 1; k < 32; ++k) mx = fmaxf(mx, row[k]);
    float ex[32]; float s = 0.f;
#pragma unroll
    for (int k = 0; k < 32; ++k) { ex[k] = __expf(row[k] - mx); s += ex[k]; }
    float inv = 1.0f / s;
    __bf16* prow = ps + tid * 32;
#pragma unroll
    for (int k = 0; k < 32; ++k) prow[k] = (__bf16)(ex[k] * inv);
  }
  __syncthreads();

  // o = p @ v : per head 2x2 tiles, K=32. 16 tiles total.
  for (int i = 0; i < 2; ++i) {
    int idx = wave * 2 + i;
    int h = idx >> 2, mt = (idx >> 1) & 1, nt = idx & 1;
    v8f acc = {};
    acc = wmma_bf(load_frag(ps + h * 1024, mt * 16, 32, 0),
                  load_frag(vts + h * 32 * 32, nt * 16, 32, 0), acc);
    int n = h * 32 + nt * 16 + (lane & 15);
    int m = mt * 16 + ((lane >> 4) << 3);
#pragma unroll
    for (int r = 0; r < 8; ++r) os[(m + r) * 128 + n] = (__bf16)acc[r];
  }
  __syncthreads();

  // y = o @ w_out^T + b_out  (M=32,N=128,K=128): 16 tiles
  for (int i = 0; i < 2; ++i) {
    int idx = wave * 2 + i;
    int mt = idx >> 3, nt = idx & 7;
    int n0 = nt * 16, m0 = mt * 16;
    v8f acc = {};
    for (int k0 = 0; k0 < 128; k0 += 32)
      acc = wmma_bf(load_frag(os, m0, 128, k0), load_frag(w_out, n0, 128, k0), acc);
    int n = n0 + (lane & 15);
    int m = m0 + ((lane >> 4) << 3);
    float bias = b_out[n];
#pragma unroll
    for (int r = 0; r < 8; ++r) ys[(m + r) * 128 + n] = acc[r] + bias;
  }
  __syncthreads();

  // mean over 32 tokens
  if (tid < 128) {
    float s = 0.f;
#pragma unroll
    for (int m = 0; m < 32; ++m) s += ys[m * 128 + tid];
    s *= (1.0f / 32.0f);
    oe_f [(size_t)bo * 128 + tid] = s;
    oe_bf[(size_t)bo * 128 + tid] = (__bf16)s;
  }
}

// ---------------------------------------------------------------------------
// Generic WMMA GEMM: C[M,N] = A[M,K](bf16) @ W[N,K]^T(bf16) + bias, opt relu.
// Block = 8 waves, tile 32x64. grid = (N/64, M/32).
// ---------------------------------------------------------------------------
__global__ __launch_bounds__(256) void gemm_kernel(
    const __bf16* __restrict__ A, int lda,
    const __bf16* __restrict__ W, int ldw,
    const float* __restrict__ bias,
    float* __restrict__ Cf, __bf16* __restrict__ Cbf,
    int K, int N, int relu)
{
  const int wave = threadIdx.x >> 5;
  const int lane = threadIdx.x & 31;
  const int row0 = blockIdx.y * 32 + (wave >> 2) * 16;
  const int col0 = blockIdx.x * 64 + (wave & 3) * 16;
  v8f acc = {};
  for (int k0 = 0; k0 < K; k0 += 32) {
    if (k0 + 64 < K) {   // global_prefetch_b8 the next-but-one K slice
      __builtin_prefetch(W + (size_t)(col0 + (lane & 15)) * ldw + k0 + 64, 0, 1);
      __builtin_prefetch(A + (size_t)(row0 + (lane & 15)) * lda + k0 + 64, 0, 1);
    }
    acc = wmma_bf(load_frag(A, row0, lda, k0), load_frag(W, col0, ldw, k0), acc);
  }
  const int n = col0 + (lane & 15);
  const int m = row0 + ((lane >> 4) << 3);
  const float b = bias ? bias[n] : 0.0f;
#pragma unroll
  for (int r = 0; r < 8; ++r) {
    float v = acc[r] + b;
    if (relu) v = fmaxf(v, 0.0f);
    size_t o = (size_t)(m + r) * (size_t)N + (size_t)n;
    if (Cf)  Cf[o]  = v;
    if (Cbf) Cbf[o] = (__bf16)v;
  }
}

// ---------------------------------------------------------------------------
// GRU step: gh = h_prev @ whh^T (WMMA, 3 gates) + gates + h update.
// gi (= x@wih^T + bih) precomputed by gemm_kernel.
// Block = 4 waves: 16 batch rows x 64 hidden cols. grid = (4, 16).
// ---------------------------------------------------------------------------
__global__ __launch_bounds__(128) void gru_step_kernel(
    const __bf16* __restrict__ hprev_bf, const float* __restrict__ hprev_f,
    const __bf16* __restrict__ whh,      const float* __restrict__ bhh,
    const float* __restrict__ gi, int t_src,
    float* __restrict__ hnew_f, __bf16* __restrict__ hnew_bf,
    float* __restrict__ g_f, __bf16* __restrict__ g_bf,
    int t_out, int col_off)
{
  const int wave = threadIdx.x >> 5, lane = threadIdx.x & 31;
  const int m0 = blockIdx.y * 16;
  const int n0 = blockIdx.x * 64 + wave * 16;
  v8f hr = {}, hz = {}, hn = {};
  for (int k0 = 0; k0 < 256; k0 += 32) {
    v16bf a = load_frag(hprev_bf, m0, 256, k0);
    hr = wmma_bf(a, load_frag(whh, n0,       256, k0), hr);
    hz = wmma_bf(a, load_frag(whh, n0 + 256, 256, k0), hz);
    hn = wmma_bf(a, load_frag(whh, n0 + 512, 256, k0), hn);
  }
  const int n = n0 + (lane & 15);
  const int m = m0 + ((lane >> 4) << 3);
  const float br = bhh[n], bz = bhh[n + 256], bn = bhh[n + 512];
#pragma unroll
  for (int r = 0; r < 8; ++r) {
    int b = m + r;
    const float* gir = gi + ((size_t)b * 32 + t_src) * 768;
    float rr = sigmoidf_(gir[n]       + hr[r] + br);
    float zz = sigmoidf_(gir[n + 256] + hz[r] + bz);
    float nn = tanhf   (gir[n + 512] + rr * (hn[r] + bn));
    float hp = hprev_f[(size_t)b * 256 + n];
    float hv = (1.0f - zz) * nn + zz * hp;
    hnew_f [(size_t)b * 256 + n] = hv;
    hnew_bf[(size_t)b * 256 + n] = (__bf16)hv;
    size_t go = ((size_t)b * 32 + t_out) * 512 + (size_t)(col_off + n);
    g_f [go] = hv;
    g_bf[go] = (__bf16)hv;
  }
}

// ---------------------------------------------------------------------------
// Transformer attention: one block per (batch, head); S=32, hd=128.
// ---------------------------------------------------------------------------
__global__ __launch_bounds__(256) void trans_attn_kernel(
    const __bf16* __restrict__ qkv,   // [8192][1536]
    __bf16* __restrict__ attn_bf)     // [8192][512]
{
  __shared__ __align__(16) __bf16 qs[32 * 128];
  __shared__ __align__(16) __bf16 ks[32 * 128];
  __shared__ __align__(16) __bf16 vts[128 * 32];
  __shared__ float sc[32 * 32];
  __shared__ __align__(16) __bf16 ps[32 * 32];
  const int b = blockIdx.x, h = blockIdx.y;
  const int tid = threadIdx.x, wave = tid >> 5, lane = tid & 31;

  // vectorized loads: 16B chunks for q/k; per-element transpose for v
  for (int i = tid; i < 32 * 16; i += 256) {
    int t = i >> 4, c8 = (i & 15) << 3;
    const __bf16* row = qkv + (size_t)(b * 32 + t) * 1536 + h * 128;
    *(v8bf*)(qs + t * 128 + c8) = *(const v8bf*)(row + c8);
    *(v8bf*)(ks + t * 128 + c8) = *(const v8bf*)(row + 512 + c8);
    v8bf vv = *(const v8bf*)(row + 1024 + c8);
#pragma unroll
    for (int j = 0; j < 8; ++j) vts[(c8 + j) * 32 + t] = vv[j];
  }
  __syncthreads();

  if (wave < 4) {   // scores: 2x2 tiles, K=128
    int mt = wave >> 1, nt = wave & 1;
    v8f acc = {};
    for (int k0 = 0; k0 < 128; k0 += 32)
      acc = wmma_bf(load_frag(qs, mt * 16, 128, k0),
                    load_frag(ks, nt * 16, 128, k0), acc);
    int n = nt * 16 + (lane & 15);
    int m = mt * 16 + ((lane >> 4) << 3);
#pragma unroll
    for (int r = 0; r < 8; ++r)
      sc[(m + r) * 32 + n] = acc[r] * 0.08838834764831845f;
  }
  __syncthreads();

  if (tid < 32) {
    const float* row = sc + tid * 32;
    float mx = row[0];
#pragma unroll
    for (int k = 1; k < 32; ++k) mx = fmaxf(mx, row[k]);
    float ex[32]; float s = 0.f;
#pragma unroll
    for (int k = 0; k < 32; ++k) { ex[k] = __expf(row[k] - mx); s += ex[k]; }
    float inv = 1.0f / s;
    __bf16* prow = ps + tid * 32;
#pragma unroll
    for (int k = 0; k < 32; ++k) prow[k] = (__bf16)(ex[k] * inv);
  }
  __syncthreads();

  // o = p @ v : 2x8 tiles, K=32
  for (int i = 0; i < 2; ++i) {
    int idx = wave * 2 + i;
    int mt = idx >> 3, nt = idx & 7;
    v8f acc = {};
    acc = wmma_bf(load_frag(ps, mt * 16, 32, 0),
                  load_frag(vts, nt * 16, 32, 0), acc);
    int n = nt * 16 + (lane & 15);
    int m = mt * 16 + ((lane >> 4) << 3);
#pragma unroll
    for (int r = 0; r < 8; ++r)
      attn_bf[(size_t)(b * 32 + m + r) * 512 + (size_t)(h * 128 + n)] = (__bf16)acc[r];
  }
}

// ---------------------------------------------------------------------------
// LayerNorm (opt residual add, opt relu). One wave per row, rows = 8*gridDim.
// ---------------------------------------------------------------------------
template <int RL>
__global__ __launch_bounds__(256) void ln_kernel(
    const float* __restrict__ x, const float* __restrict__ res,
    const float* __restrict__ gam, const float* __restrict__ bet,
    float* __restrict__ outf, __bf16* __restrict__ outbf, int relu)
{
  const int row  = blockIdx.x * 8 + (threadIdx.x >> 5);
  const int lane = threadIdx.x & 31;
  constexpr int PER = RL / 32;
  const float* xr = x + (size_t)row * RL;
  float v[PER];
  float s = 0.f;
#pragma unroll
  for (int i = 0; i < PER; ++i) {
    int c = lane + i * 32;
    float t = xr[c];
    if (res) t += res[(size_t)row * RL + c];
    v[i] = t; s += t;
  }
#pragma unroll
  for (int o = 16; o > 0; o >>= 1) s += __shfl_xor(s, o, 32);
  float mean = s * (1.0f / RL);
  float q = 0.f;
#pragma unroll
  for (int i = 0; i < PER; ++i) { float d = v[i] - mean; q += d * d; }
#pragma unroll
  for (int o = 16; o > 0; o >>= 1) q += __shfl_xor(q, o, 32);
  float inv = rsqrtf(q * (1.0f / RL) + 1e-5f);
#pragma unroll
  for (int i = 0; i < PER; ++i) {
    int c = lane + i * 32;
    float o = (v[i] - mean) * inv * gam[c] + bet[c];
    if (relu) o = fmaxf(o, 0.f);
    if (outf)  outf [(size_t)row * RL + c] = o;
    if (outbf) outbf[(size_t)row * RL + c] = (__bf16)o;
  }
}

// ---------------------------------------------------------------------------
// Mean over sequence + tiny temporal MLP (2->64->128) + concat (bf16).
// One block per batch element.
// ---------------------------------------------------------------------------
__global__ __launch_bounds__(256) void mean_temp_concat_kernel(
    const float* __restrict__ g2,                // [8192][512]
    const float* __restrict__ days_between,      // [256][31]
    const float* __restrict__ days_since,        // [256]
    const float* __restrict__ te1_w, const float* __restrict__ te1_b,
    const float* __restrict__ ln1g,  const float* __restrict__ ln1b,
    const float* __restrict__ te2_w, const float* __restrict__ te2_b,
    const float* __restrict__ ln2g,  const float* __restrict__ ln2b,
    __bf16* __restrict__ concat_bf)              // [256][640]
{
  const int b = blockIdx.x, tid = threadIdx.x;
  for (int c = tid; c < 512; c += 256) {
    float s = 0.f;
    for (int t = 0; t < 32; ++t) s += g2[((size_t)b * 32 + t) * 512 + c];
    concat_bf[(size_t)b * 640 + c] = (__bf16)(s * (1.0f / 32.0f));
  }
  __shared__ float sh[192];
  __shared__ float st[4];
  if (tid == 0) {
    float s = 0.f;
    for (int t = 0; t < 31; ++t) s += days_between[b * 31 + t];
    st[0] = s * (1.0f / 31.0f);
    st[1] = days_since[b];
  }
  __syncthreads();
  if (tid < 64) sh[tid] = te1_w[tid * 2] * st[0] + te1_w[tid * 2 + 1] * st[1] + te1_b[tid];
  __syncthreads();
  if (tid == 0) {
    float m = 0.f; for (int i = 0; i < 64; ++i) m += sh[i]; m *= (1.f / 64.f);
    float q = 0.f; for (int i = 0; i < 64; ++i) { float d = sh[i] - m; q += d * d; }
    st[2] = m; st[3] = rsqrtf(q * (1.f / 64.f) + 1e-5f);
  }
  __syncthreads();
  if (tid < 64) sh[tid] = fmaxf((sh[tid] - st[2]) * st[3] * ln1g[tid] + ln1b[tid], 0.f);
  __syncthreads();
  if (tid < 128) {
    float s = te2_b[tid];
    for (int j = 0; j < 64; ++j) s += te2_w[tid * 64 + j] * sh[j];
    sh[64 + tid] = s;
  }
  __syncthreads();
  if (tid == 0) {
    float m = 0.f; for (int i = 0; i < 128; ++i) m += sh[64 + i]; m *= (1.f / 128.f);
    float q = 0.f; for (int i = 0; i < 128; ++i) { float d = sh[64 + i] - m; q += d * d; }
    st[2] = m; st[3] = rsqrtf(q * (1.f / 128.f) + 1e-5f);
  }
  __syncthreads();
  if (tid < 128) {
    float o = (sh[64 + tid] - st[2]) * st[3] * ln2g[tid] + ln2b[tid];
    concat_bf[(size_t)b * 640 + 512 + tid] = (__bf16)fmaxf(o, 0.f);
  }
}

// ---------------------------------------------------------------------------
// Host orchestration.  Workspace budget ~131 MB (bf16 weights 6MB, persistent
// activations ~75MB, 50MB arena time-shared: GRU gi -> transformer qkv/attn ->
// FF intermediates).  No atomics -> bitwise deterministic.  Graph-capture safe.
// ---------------------------------------------------------------------------
extern "C" void kernel_launch(void* const* d_in, const int* in_sizes, int n_in,
                              void* d_out, int out_size, void* d_ws, size_t ws_size,
                              hipStream_t stream) {
  (void)in_sizes; (void)n_in; (void)out_size; (void)ws_size;
  char* ws = (char*)d_ws;
  size_t off = 0;
  auto take = [&](size_t bytes) {
    size_t r = off; off += (bytes + 255) & ~(size_t)255; return r;
  };
  // bf16 weights
  __bf16* w_ain  = (__bf16*)(ws + take(384 * 128 * 2));
  __bf16* w_aout = (__bf16*)(ws + take(128 * 128 * 2));
  __bf16* w_wihf = (__bf16*)(ws + take(768 * 128 * 2));
  __bf16* w_whhf = (__bf16*)(ws + take(768 * 256 * 2));
  __bf16* w_wihb = (__bf16*)(ws + take(768 * 128 * 2));
  __bf16* w_whhb = (__bf16*)(ws + take(768 * 256 * 2));
  __bf16* w_tin  = (__bf16*)(ws + take(1536 * 512 * 2));
  __bf16* w_tout = (__bf16*)(ws + take(512 * 512 * 2));
  __bf16* w_ff1  = (__bf16*)(ws + take(1024 * 512 * 2));
  __bf16* w_ff2  = (__bf16*)(ws + take(512 * 1024 * 2));
  __bf16* w_c    = (__bf16*)(ws + take(256 * 640 * 2));
  // persistent activations
  float*  oe_f   = (float*) (ws + take(8192 * 128 * 4));
  __bf16* oe_bf  = (__bf16*)(ws + take(8192 * 128 * 2));
  float*  h_f0   = (float*) (ws + take(256 * 256 * 4));
  float*  h_f1   = (float*) (ws + take(256 * 256 * 4));
  __bf16* h_bf0  = (__bf16*)(ws + take(256 * 256 * 2));
  __bf16* h_bf1  = (__bf16*)(ws + take(256 * 256 * 2));
  float*  g_f    = (float*) (ws + take(8192 * 512 * 4));
  __bf16* g_bf   = (__bf16*)(ws + take(8192 * 512 * 2));
  float*  g1_f   = (float*) (ws + take(8192 * 512 * 4));
  __bf16* g1_bf  = (__bf16*)(ws + take(8192 * 512 * 2));
  float*  g2_f   = (float*) (ws + take(8192 * 512 * 4));
  __bf16* cat_bf = (__bf16*)(ws + take(256 * 640 * 2));
  float*  cls_f  = (float*) (ws + take(256 * 256 * 4));
  // time-shared arena (50,331,648 bytes)
  char* arena = ws + take(50331648);
  float*  gi_f    = (float*) (arena);
  float*  gi_b    = (float*) (arena + 25165824);
  __bf16* qkv_bf  = (__bf16*)(arena);
  __bf16* attn_bf = (__bf16*)(arena + 25165824);
  float*  atto_f  = (float*) (arena + 33554432);
  __bf16* relu_bf = (__bf16*)(arena);
  float*  ff_f    = (float*) (arena + 16777216);

  auto F = [&](int i) { return (const float*)d_in[i]; };
  auto cvt = [&](int idx, __bf16* dst, int n) {   // n multiple of 4 for all weights
    int n4 = n / 4;
    cvt_f32_bf16_kernel<<<(n4 + 255) / 256, 256, 0, stream>>>(F(idx), dst, n4);
  };
  cvt(4,  w_ain, 49152);   cvt(6,  w_aout, 16384);
  cvt(8,  w_wihf, 98304);  cvt(9,  w_whhf, 196608);
  cvt(12, w_wihb, 98304);  cvt(13, w_whhb, 196608);
  cvt(16, w_tin, 786432);  cvt(18, w_tout, 262144);
  cvt(22, w_ff1, 524288);  cvt(24, w_ff2, 524288);
  cvt(36, w_c, 163840);

  // 1) item self-attention + mean pool
  attn_pool_kernel<<<8192, 256, 0, stream>>>(
      (const int*)d_in[0], F(3), w_ain, F(5), w_aout, F(7), oe_f, oe_bf);

  // 2) GRU input-side GEMMs (gi = x @ wih^T + bih), both directions
  gemm_kernel<<<dim3(12, 256), 256, 0, stream>>>(oe_bf, 128, w_wihf, 128, F(10),
                                                 gi_f, nullptr, 128, 768, 0);
  gemm_kernel<<<dim3(12, 256), 256, 0, stream>>>(oe_bf, 128, w_wihb, 128, F(14),
                                                 gi_b, nullptr, 128, 768, 0);

  // 3) GRU forward (ping-pong h buffers)
  float*  hf[2]  = {h_f0, h_f1};
  __bf16* hbf[2] = {h_bf0, h_bf1};
  zero_u32_kernel<<<256, 256, 0, stream>>>((uint32_t*)h_f0, 65536);
  zero_u32_kernel<<<128, 256, 0, stream>>>((uint32_t*)h_bf0, 32768);
  for (int s = 0; s < 32; ++s) {
    int p = s & 1;
    gru_step_kernel<<<dim3(4, 16), 128, 0, stream>>>(
        hbf[p], hf[p], w_whhf, F(11), gi_f, s,
        hf[p ^ 1], hbf[p ^ 1], g_f, g_bf, s, 0);
  }
  // 4) GRU backward (reversed sequence; write to unreversed positions)
  zero_u32_kernel<<<256, 256, 0, stream>>>((uint32_t*)h_f0, 65536);
  zero_u32_kernel<<<128, 256, 0, stream>>>((uint32_t*)h_bf0, 32768);
  for (int s = 0; s < 32; ++s) {
    int p = s & 1;
    gru_step_kernel<<<dim3(4, 16), 128, 0, stream>>>(
        hbf[p], hf[p], w_whhb, F(15), gi_b, 31 - s,
        hf[p ^ 1], hbf[p ^ 1], g_f, g_bf, 31 - s, 256);
  }

  // 5) transformer encoder layer (post-LN)
  gemm_kernel<<<dim3(24, 256), 256, 0, stream>>>(g_bf, 512, w_tin, 512, F(17),
                                                 nullptr, qkv_bf, 512, 1536, 0);
  trans_attn_kernel<<<dim3(256, 4), 256, 0, stream>>>(qkv_bf, attn_bf);
  gemm_kernel<<<dim3(8, 256), 256, 0, stream>>>(attn_bf, 512, w_tout, 512, F(19),
                                                atto_f, nullptr, 512, 512, 0);
  ln_kernel<512><<<1024, 256, 0, stream>>>(g_f, atto_f, F(20), F(21), g1_f, g1_bf, 0);
  gemm_kernel<<<dim3(16, 256), 256, 0, stream>>>(g1_bf, 512, w_ff1, 512, F(23),
                                                 nullptr, relu_bf, 512, 1024, 1);
  gemm_kernel<<<dim3(8, 256), 256, 0, stream>>>(relu_bf, 1024, w_ff2, 1024, F(25),
                                                ff_f, nullptr, 1024, 512, 0);
  ln_kernel<512><<<1024, 256, 0, stream>>>(g1_f, ff_f, F(26), F(27), g2_f, nullptr, 0);

  // 6) sequence mean + temporal MLP + concat
  mean_temp_concat_kernel<<<256, 256, 0, stream>>>(
      g2_f, F(1), F(2), F(28), F(29), F(30), F(31), F(32), F(33), F(34), F(35),
      cat_bf);

  // 7) classifier + final LN + relu -> d_out
  gemm_kernel<<<dim3(4, 8), 256, 0, stream>>>(cat_bf, 640, w_c, 640, F(37),
                                              cls_f, nullptr, 640, 256, 0);
  ln_kernel<256><<<32, 256, 0, stream>>>(cls_f, nullptr, F(38), F(39),
                                         (float*)d_out, nullptr, 1);
}
